// TransformerWindowClassifier_9981503995983
// MI455X (gfx1250) — matrix operand order repaired and checked by reading, not compiled, not run on hardware
//
#include <hip/hip_runtime.h>
#include <hip/hip_bf16.h>
#include <cstdint>

// ---------------------------------------------------------------------------
// TransformerWindowClassifier for MI455X (gfx1250, wave32, WMMA).
// All GEMMs run on v_wmma_f32_16x16x32_bf16 (bf16 in, fp32 accumulate).
// Block tile 64x128, 8 waves, 32x32 per wave (4 WMMAs / wave / K-chunk),
// double-buffered LDS staging via GLOBAL_LOAD_ASYNC_TO_LDS_B128 (ASYNCcnt),
// one barrier per K-chunk.
// ---------------------------------------------------------------------------

#define B_   32
#define S_   512
#define D_   512
#define H_   8
#define HD_  64
#define DFF_ 2048
#define KIN_ 128
#define W_   511
#define NC_  1000
#define L_   4

// LDS row stride: 40 bf16 = 80 bytes -> 16B-aligned rows (async B128 dest)
// and conflict-free fragment reads (row*20 mod 64 distinct over 16 rows).
#define LDSK 40

typedef __attribute__((ext_vector_type(16))) __bf16 v16bf;
typedef __attribute__((ext_vector_type(8)))  float  v8f;

union FragBF { v16bf v; uint32_t u[8]; };

__device__ __forceinline__ uint32_t lds_addr32(const void* p)
{
    // generic pointer to a __shared__ object: low 32 bits are the LDS offset
    return (uint32_t)(uintptr_t)p;
}

__device__ __forceinline__ void async_b128(uint32_t lds_dst, const void* gsrc)
{
    asm volatile("global_load_async_to_lds_b128 %0, %1, off"
                 :: "v"(lds_dst), "v"((uint64_t)(uintptr_t)gsrc) : "memory");
}

__device__ __forceinline__ void wait_async0()
{
    asm volatile("s_wait_asynccnt 0x0" ::: "memory");
}

// ---------------------------------------------------------------------------
// Generic WMMA GEMM:  C[M,N] = epi( scale * A[M,K] @ W + bias )  (+ residual)
//   A: bf16 row-major, lda elements, per-z offset sAz.
//   W: bf16. TRANSB=1: W is (N x K) row-major (pre-transposed weights, and
//            K-matrix for Q@K^T) -> async 16B loads straight into LDS.
//            TRANSB=0: W is (K x N) row-major (only P@V uses this).
//   Output: fp32 (Cf) or bf16 (Cb), ldc elements, per-z offset sCz.
//   Residual (fp32) shares C indexing.
// Requires K % 32 == 0; 16B-aligned A/W rows (all call sites satisfy this).
// M and N are handled by row/col clamping on loads + guards on stores.
// ---------------------------------------------------------------------------
template<bool TRANSB, bool HAS_BIAS, bool GELU, bool RESID, bool OUT_BF16>
__global__ __launch_bounds__(256) void gemm_wmma(
    int M, int N, int K,
    const __bf16* __restrict__ A, int lda, long long sAz, long long offA0,
    const __bf16* __restrict__ Wp, int ldw, long long sWz, long long offW0,
    const float* __restrict__ bias, float scale,
    float* __restrict__ Cf, __bf16* __restrict__ Cb,
    int ldc, long long sCz, long long offC0,
    const float* __restrict__ resid)
{
    __shared__ __bf16 As[2][64][LDSK];
    __shared__ __bf16 Bs[2][128][LDSK];   // n-major: Bs[buf][n][k]

    const int tid   = threadIdx.x;
    const int wave  = tid >> 5;
    const int lane  = tid & 31;
    const int wm    = wave >> 2;        // 0..1 -> 32-row slice
    const int wn    = wave & 3;         // 0..3 -> 32-col slice
    const int lhalf = lane >> 4;        // 0/1
    const int lmod  = lane & 15;

    const int m0 = blockIdx.y * 64;
    const int n0 = blockIdx.x * 128;

    const __bf16* Ab = A  + offA0 + (long long)blockIdx.z * sAz;
    const __bf16* Wb = Wp + offW0 + (long long)blockIdx.z * sWz;

    // staging coordinates
    const int a_m = tid >> 2;           // 0..63
    const int a_k = (tid & 3) * 8;      // 0,8,16,24
    const int b_n = tid >> 1;           // 0..127 (TRANSB)
    const int b_k = (tid & 1) * 16;     // 0/16   (TRANSB)
    const int c_n = tid & 127;          // 0..127 (non-TRANSB)
    const int c_k = (tid >> 7) * 16;    // 0/16

    // clamped source rows: OOB A rows only feed discarded C rows; OOB B rows
    // (n-major) only feed discarded C cols.
    const int a_mg = (m0 + a_m < M) ? (m0 + a_m) : (M - 1);
    const int b_ng = (n0 + b_n < N) ? (n0 + b_n) : (N - 1);

    v8f zero = {0.f,0.f,0.f,0.f,0.f,0.f,0.f,0.f};
    v8f acc[2][2];
    acc[0][0] = zero; acc[0][1] = zero; acc[1][0] = zero; acc[1][1] = zero;

    auto fill = [&](int k0, int buf) {
        // ---- A tile: 64 x 32, one async B128 per thread ----
        async_b128(lds_addr32(&As[buf][a_m][a_k]),
                   Ab + (size_t)a_mg * lda + k0 + a_k);
        // ---- B tile: 32(k) x 128(n), stored n-major ----
        if (TRANSB) {
            const __bf16* src = Wb + (size_t)b_ng * ldw + k0 + b_k;
            const uint32_t dst = lds_addr32(&Bs[buf][b_n][b_k]);
            async_b128(dst,      src);
            async_b128(dst + 16, src + 8);
        } else {
            const int ng = n0 + c_n;
            if (ng < N) {
                #pragma unroll
                for (int kk = 0; kk < 16; ++kk)
                    Bs[buf][c_n][c_k + kk] = Wb[(size_t)(k0 + c_k + kk) * ldw + ng];
            } else {
                #pragma unroll
                for (int kk = 0; kk < 16; ++kk)
                    Bs[buf][c_n][c_k + kk] = (__bf16)0.0f;
            }
        }
    };

    auto compute = [&](int buf) {
        FragBF af[2], bfg[2];
        #pragma unroll
        for (int i = 0; i < 2; ++i) {
            const __bf16* ar = &As[buf][wm * 32 + i * 16 + lmod][0];
            #pragma unroll
            for (int j = 0; j < 8; ++j) {
                const int kj = (j < 4 ? 2 * j : 16 + 2 * (j - 4)) + lhalf * 8;
                af[i].u[j] = *reinterpret_cast<const uint32_t*>(ar + kj);
            }
        }
        #pragma unroll
        for (int i = 0; i < 2; ++i) {
            const __bf16* br = &Bs[buf][wn * 32 + i * 16 + lmod][0];
            #pragma unroll
            for (int j = 0; j < 8; ++j) {
                const int kj = (j < 4 ? 2 * j : 16 + 2 * (j - 4)) + lhalf * 8;
                bfg[i].u[j] = *reinterpret_cast<const uint32_t*>(br + kj);
            }
        }
        #pragma unroll
        for (int i = 0; i < 2; ++i)
            #pragma unroll
            for (int j = 0; j < 2; ++j)
                acc[i][j] = __builtin_amdgcn_wmma_f32_16x16x32_bf16(
                    false, af[i].v, false, bfg[j].v, (short)0, acc[i][j],
                    false, false);
    };

    const int nch = K >> 5;
    fill(0, 0);
    wait_async0();
    __syncthreads();
    for (int i = 0; i < nch; ++i) {
        if (i + 1 < nch) fill((i + 1) << 5, (i + 1) & 1);  // overlap with WMMAs
        compute(i & 1);
        wait_async0();
        __syncthreads();
    }

    // ---- epilogue: C layout -> lane(n = lmod), vgpr r (m = lhalf*8 + r) ----
    const long long offC = offC0 + (long long)blockIdx.z * sCz;
    #pragma unroll
    for (int i = 0; i < 2; ++i) {
        #pragma unroll
        for (int j = 0; j < 2; ++j) {
            const int n = n0 + wn * 32 + j * 16 + lmod;
            if (n < N) {
                #pragma unroll
                for (int r = 0; r < 8; ++r) {
                    const int m = m0 + wm * 32 + i * 16 + lhalf * 8 + r;
                    if (m < M) {
                        float v = acc[i][j][r] * scale;
                        if (HAS_BIAS) v += bias[n];
                        if (GELU)     v = 0.5f * v * (1.0f + erff(v * 0.70710678118f));
                        const size_t idx = (size_t)offC + (size_t)m * ldc + n;
                        if (RESID)    v += resid[idx];
                        if (OUT_BF16) Cb[idx] = (__bf16)v;
                        else          Cf[idx] = v;
                    }
                }
            }
        }
    }
}

// ---------------------------------------------------------------------------
// fp32 -> bf16 convert (optional device-side scale from a 1-elem array)
// ---------------------------------------------------------------------------
__global__ void cvt_bf16_kernel(const float* __restrict__ x, __bf16* __restrict__ y,
                                const float* __restrict__ scale_ptr, int n)
{
    const int i = blockIdx.x * 256 + threadIdx.x;
    if (i < n) {
        const float s = scale_ptr ? scale_ptr[0] : 1.0f;
        y[i] = (__bf16)(x[i] * s);
    }
}

// ---------------------------------------------------------------------------
// Transposing convert: x (Lc, K, N) fp32 -> y (Lc, N, K) bf16
// ---------------------------------------------------------------------------
__global__ void cvtT_bf16_kernel(const float* __restrict__ x, __bf16* __restrict__ y,
                                 int Lc, int Kd, int Nd)
{
    const int i = blockIdx.x * 256 + threadIdx.x;
    const int total = Lc * Kd * Nd;
    if (i < total) {
        const int l = i / (Kd * Nd);
        const int rem = i - l * (Kd * Nd);
        const int n = rem / Kd;
        const int k = rem - n * Kd;
        y[i] = (__bf16)x[(size_t)l * Kd * Nd + (size_t)k * Nd + n];
    }
}

// ---------------------------------------------------------------------------
// h[b,0,:] = cls + pos[0];  h[b,s,:] = proj[b,s-1,:] + pos[s]   (fp32)
// ---------------------------------------------------------------------------
__global__ void embed_assemble_kernel(const float* __restrict__ proj,
                                      const float* __restrict__ cls,
                                      const float* __restrict__ pos,
                                      float* __restrict__ h)
{
    const int idx = blockIdx.x * 256 + threadIdx.x;   // over B*S*D
    const int d = idx & (D_ - 1);
    const int s = (idx >> 9) & (S_ - 1);
    const int b = idx >> 18;
    float v = pos[s * D_ + d];
    if (s == 0) v += cls[d];
    else        v += proj[((size_t)b * W_ + (s - 1)) * D_ + d];
    h[idx] = v;
}

// ---------------------------------------------------------------------------
// LayerNorm over D=512, one 256-thread block per row, bf16 output
// ---------------------------------------------------------------------------
__global__ void layernorm512_kernel(const float* __restrict__ h,
                                    const float* __restrict__ g,
                                    const float* __restrict__ bta,
                                    __bf16* __restrict__ o)
{
    __shared__ float red[256];
    const int row = blockIdx.x, t = threadIdx.x;
    const float* hr = h + (size_t)row * 512;
    const float a = hr[t], c = hr[t + 256];

    red[t] = a + c; __syncthreads();
    for (int st = 128; st > 0; st >>= 1) { if (t < st) red[t] += red[t + st]; __syncthreads(); }
    const float mean = red[0] * (1.0f / 512.0f);
    __syncthreads();

    const float da = a - mean, dc = c - mean;
    red[t] = da * da + dc * dc; __syncthreads();
    for (int st = 128; st > 0; st >>= 1) { if (t < st) red[t] += red[t + st]; __syncthreads(); }
    const float rstd = rsqrtf(red[0] * (1.0f / 512.0f) + 1e-5f);

    __bf16* orow = o + (size_t)row * 512;
    orow[t]       = (__bf16)(da * rstd * g[t]       + bta[t]);
    orow[t + 256] = (__bf16)(dc * rstd * g[t + 256] + bta[t + 256]);
}

// ---------------------------------------------------------------------------
// Softmax over 512, one 256-thread block per row, bf16 output
// ---------------------------------------------------------------------------
__global__ void softmax512_kernel(const float* __restrict__ s, __bf16* __restrict__ p)
{
    __shared__ float red[256];
    const int row = blockIdx.x, t = threadIdx.x;
    const float* sr = s + (size_t)row * 512;
    const float a = sr[t], b = sr[t + 256];

    red[t] = fmaxf(a, b); __syncthreads();
    for (int st = 128; st > 0; st >>= 1) { if (t < st) red[t] = fmaxf(red[t], red[t + st]); __syncthreads(); }
    const float mx = red[0];
    __syncthreads();

    const float ea = __expf(a - mx), eb = __expf(b - mx);
    red[t] = ea + eb; __syncthreads();
    for (int st = 128; st > 0; st >>= 1) { if (t < st) red[t] += red[t + st]; __syncthreads(); }
    const float inv = 1.0f / red[0];

    __bf16* pr = p + (size_t)row * 512;
    pr[t]       = (__bf16)(ea * inv);
    pr[t + 256] = (__bf16)(eb * inv);
}

// ---------------------------------------------------------------------------
// cls_bf[b,:] = bf16(h[b,0,:])
// ---------------------------------------------------------------------------
__global__ void extract_cls_kernel(const float* __restrict__ h, __bf16* __restrict__ c)
{
    const int i = blockIdx.x * 256 + threadIdx.x;   // over B*D
    const int b = i >> 9, d = i & (D_ - 1);
    c[i] = (__bf16)h[(size_t)b * S_ * D_ + d];
}

// ---------------------------------------------------------------------------
// Host orchestration
// ---------------------------------------------------------------------------
static inline size_t align256(size_t x) { return (x + 255) & ~(size_t)255; }

extern "C" void kernel_launch(void* const* d_in, const int* in_sizes, int n_in,
                              void* d_out, int out_size, void* d_ws, size_t ws_size,
                              hipStream_t stream)
{
    (void)in_sizes; (void)n_in; (void)out_size; (void)ws_size;

    const float* x_seq     = (const float*)d_in[0];
    const float* in_scale  = (const float*)d_in[1];
    const float* in_proj_w = (const float*)d_in[2];
    const float* in_proj_b = (const float*)d_in[3];
    const float* cls       = (const float*)d_in[4];
    const float* pos       = (const float*)d_in[5];
    const float* ln1_g     = (const float*)d_in[6];
    const float* ln1_b     = (const float*)d_in[7];
    const float* qkv_w     = (const float*)d_in[8];
    const float* qkv_b     = (const float*)d_in[9];
    const float* out_w     = (const float*)d_in[10];
    const float* out_b     = (const float*)d_in[11];
    const float* ln2_g     = (const float*)d_in[12];
    const float* ln2_b     = (const float*)d_in[13];
    const float* ffn_w1    = (const float*)d_in[14];
    const float* ffn_b1    = (const float*)d_in[15];
    const float* ffn_w2    = (const float*)d_in[16];
    const float* ffn_b2    = (const float*)d_in[17];
    const float* head_w1   = (const float*)d_in[18];
    const float* head_b1   = (const float*)d_in[19];
    const float* head_w2   = (const float*)d_in[20];
    const float* head_b2   = (const float*)d_in[21];
    float* logits = (float*)d_out;

    // ---- workspace bump allocator ----
    char* ws = (char*)d_ws;
    size_t off = 0;
    auto bump = [&](size_t bytes) -> void* {
        void* p = ws + off; off = align256(off + bytes); return p;
    };

    const size_t BS = (size_t)B_ * S_;           // 16384
    const size_t BW = (size_t)B_ * W_;           // 16352

    __bf16* inw_bf  = (__bf16*)bump((size_t)KIN_ * D_ * 2);        // (D, KIN) transposed
    __bf16* qkvw_bf = (__bf16*)bump((size_t)L_ * D_ * 3 * D_ * 2); // (3D, D) per layer
    __bf16* outw_bf = (__bf16*)bump((size_t)L_ * D_ * D_ * 2);
    __bf16* f1w_bf  = (__bf16*)bump((size_t)L_ * D_ * DFF_ * 2);
    __bf16* f2w_bf  = (__bf16*)bump((size_t)L_ * DFF_ * D_ * 2);
    __bf16* h1w_bf  = (__bf16*)bump((size_t)D_ * D_ * 2);
    __bf16* h2w_bf  = (__bf16*)bump((size_t)D_ * NC_ * 2);         // (NC, D)
    __bf16* xsc_bf  = (__bf16*)bump(BW * KIN_ * 2);
    float*  hbuf    = (float*)bump(BS * D_ * 4);
    __bf16* xln_bf  = (__bf16*)bump(BS * D_ * 2);
    __bf16* qkv_bf  = (__bf16*)bump(BS * 3 * D_ * 2);
    float*  scores  = (float*)bump(BS * S_ * 4);   // per-head; reused as proj
    __bf16* p_bf    = (__bf16*)bump(BS * S_ * 2);
    __bf16* o_bf    = (__bf16*)bump(BS * D_ * 2);
    __bf16* f_bf    = (__bf16*)bump(BS * DFF_ * 2);
    __bf16* cls_bf  = (__bf16*)bump((size_t)B_ * D_ * 2);
    __bf16* hid_bf  = (__bf16*)bump((size_t)B_ * D_ * 2);
    float*  proj_f  = scores;  // alias: proj is dead before attention starts

    #define CVT_T(src, dst, Lc, Kd, Nd) \
        cvtT_bf16_kernel<<<((Lc) * (Kd) * (Nd) + 255) / 256, 256, 0, stream>>>((src), (dst), (Lc), (Kd), (Nd))

    // ---- weights -> bf16, transposed to (N, K) for contiguous TRANSB loads ----
    CVT_T(in_proj_w, inw_bf,  1, KIN_, D_);
    CVT_T(qkv_w,     qkvw_bf, L_, D_, 3 * D_);
    CVT_T(out_w,     outw_bf, L_, D_, D_);
    CVT_T(ffn_w1,    f1w_bf,  L_, D_, DFF_);
    CVT_T(ffn_w2,    f2w_bf,  L_, DFF_, D_);
    CVT_T(head_w1,   h1w_bf,  1, D_, D_);
    CVT_T(head_w2,   h2w_bf,  1, D_, NC_);
    // x * in_scale -> bf16
    cvt_bf16_kernel<<<((int)(BW * KIN_) + 255) / 256, 256, 0, stream>>>(
        x_seq, xsc_bf, in_scale, (int)(BW * KIN_));

    // ---- input projection: (B*W, 128) @ (128, 512)^T-stored + b -> fp32 ----
    gemm_wmma<true, true, false, false, false><<<dim3(D_ / 128, (int)((BW + 63) / 64), 1), 256, 0, stream>>>(
        (int)BW, D_, KIN_,
        xsc_bf, KIN_, 0, 0,
        inw_bf, KIN_, 0, 0,
        in_proj_b, 1.0f,
        proj_f, nullptr, D_, 0, 0, nullptr);

    // ---- assemble h = [cls; proj] + pos ----
    embed_assemble_kernel<<<(int)(BS * D_ / 256), 256, 0, stream>>>(proj_f, cls, pos, hbuf);

    const float attn_scale = 0.125f;   // 1/sqrt(64)

    for (int l = 0; l < L_; ++l) {
        // LN1 -> bf16
        layernorm512_kernel<<<(int)BS, 256, 0, stream>>>(
            hbuf, ln1_g + l * D_, ln1_b + l * D_, xln_bf);

        // QKV: (B*S, 512) @ W^T(1536,512) + b -> bf16
        gemm_wmma<true, true, false, false, true><<<dim3(3 * D_ / 128, (int)(BS / 64), 1), 256, 0, stream>>>(
            (int)BS, 3 * D_, D_,
            xln_bf, D_, 0, 0,
            qkvw_bf + (size_t)l * D_ * 3 * D_, D_, 0, 0,
            qkv_b + l * 3 * D_, 1.0f,
            nullptr, qkv_bf, 3 * D_, 0, 0, nullptr);

        // attention, one head at a time (bounds scratch)
        for (int hh = 0; hh < H_; ++hh) {
            // scores[b] = scale * Q_h @ K_h^T   (M=N=512, K=64, z=B)
            gemm_wmma<true, false, false, false, false><<<dim3(S_ / 128, S_ / 64, B_), 256, 0, stream>>>(
                S_, S_, HD_,
                qkv_bf, 3 * D_, (long long)S_ * 3 * D_, (long long)hh * HD_,
                qkv_bf, 3 * D_, (long long)S_ * 3 * D_, (long long)D_ + hh * HD_,
                nullptr, attn_scale,
                scores, nullptr, S_, (long long)S_ * S_, 0, nullptr);

            softmax512_kernel<<<(int)BS, 256, 0, stream>>>(scores, p_bf);

            // O_h[b] = P @ V_h   (M=512, N=64, K=512, z=B) -> bf16 slice of O
            gemm_wmma<false, false, false, false, true><<<dim3(1, S_ / 64, B_), 256, 0, stream>>>(
                S_, HD_, S_,
                p_bf, S_, (long long)S_ * S_, 0,
                qkv_bf, 3 * D_, (long long)S_ * 3 * D_, (long long)2 * D_ + hh * HD_,
                nullptr, 1.0f,
                nullptr, o_bf, D_, (long long)S_ * D_, (long long)hh * HD_, nullptr);
        }

        // out projection + residual: h += O @ out_w + b   (fp32)
        gemm_wmma<true, true, false, true, false><<<dim3(D_ / 128, (int)(BS / 64), 1), 256, 0, stream>>>(
            (int)BS, D_, D_,
            o_bf, D_, 0, 0,
            outw_bf + (size_t)l * D_ * D_, D_, 0, 0,
            out_b + l * D_, 1.0f,
            hbuf, nullptr, D_, 0, 0, hbuf);

        // LN2 -> bf16
        layernorm512_kernel<<<(int)BS, 256, 0, stream>>>(
            hbuf, ln2_g + l * D_, ln2_b + l * D_, xln_bf);

        // FFN1 + exact GELU -> bf16
        gemm_wmma<true, true, true, false, true><<<dim3(DFF_ / 128, (int)(BS / 64), 1), 256, 0, stream>>>(
            (int)BS, DFF_, D_,
            xln_bf, D_, 0, 0,
            f1w_bf + (size_t)l * D_ * DFF_, D_, 0, 0,
            ffn_b1 + l * DFF_, 1.0f,
            nullptr, f_bf, DFF_, 0, 0, nullptr);

        // FFN2 + residual into h (fp32)
        gemm_wmma<true, true, false, true, false><<<dim3(D_ / 128, (int)(BS / 64), 1), 256, 0, stream>>>(
            (int)BS, D_, DFF_,
            f_bf, DFF_, 0, 0,
            f2w_bf + (size_t)l * DFF_ * D_, DFF_, 0, 0,
            ffn_b2 + l * D_, 1.0f,
            hbuf, nullptr, D_, 0, 0, hbuf);
    }

    // ---- head ----
    extract_cls_kernel<<<(B_ * D_) / 256, 256, 0, stream>>>(hbuf, cls_bf);

    // hid = gelu(cls @ head_w1 + b1) -> bf16     (M=32)
    gemm_wmma<true, true, true, false, true><<<dim3(D_ / 128, 1, 1), 256, 0, stream>>>(
        B_, D_, D_,
        cls_bf, D_, 0, 0,
        h1w_bf, D_, 0, 0,
        head_b1, 1.0f,
        nullptr, hid_bf, D_, 0, 0, nullptr);

    // logits = hid @ head_w2 + b2 -> fp32 out    (N=1000, guarded)
    gemm_wmma<true, true, false, false, false><<<dim3((NC_ + 127) / 128, 1, 1), 256, 0, stream>>>(
        B_, NC_, D_,
        hid_bf, D_, 0, 0,
        h2w_bf, D_, 0, 0,
        head_b2, 1.0f,
        logits, nullptr, NC_, 0, 0, nullptr);

    #undef CVT_T
}
